// MyEncoder_38122129719316
// MI455X (gfx1250) — matrix operand order repaired and checked by reading, not compiled, not run on hardware
//
#include <hip/hip_runtime.h>
#include <hip/hip_bf16.h>
#include <cstdint>
#include <cstddef>

// Problem constants (match reference)
#define B_   32
#define T_   1000
#define F_   80
#define H_   512
#define NWG  16     // workgroups in the scan kernel (one WGP each)
#define TPB  256    // 8 wave32 waves per workgroup

typedef __attribute__((ext_vector_type(16))) __bf16 v16bf;
typedef __attribute__((ext_vector_type(8)))  float  v8f;

__device__ __forceinline__ v8f wmma_bf16(v16bf a, v16bf b, v8f c) {
  // D = A(16x32 bf16) * B(32x16 bf16) + C(16x16 f32)
  return __builtin_amdgcn_wmma_f32_16x16x32_bf16(false, a, false, b, (short)0, c,
                                                 false, false);
}

__device__ __forceinline__ float sigf(float x) { return 1.f / (1.f + __expf(-x)); }

// Async 16B copy global -> LDS (per lane), tracked on ASYNCcnt.
// dst_lds_off = LDS byte offset (low 32 bits of the flat shared address).
__device__ __forceinline__ void async_b128_to_lds(unsigned dst_lds_off, const void* gsrc) {
  asm volatile("global_load_async_to_lds_b128 %0, %1, off"
               :: "v"(dst_lds_off), "v"((unsigned long long)(uintptr_t)gsrc)
               : "memory");
}
__device__ __forceinline__ void wait_asynccnt0() {
  asm volatile("s_wait_asynccnt 0" ::: "memory");
}
__device__ __forceinline__ unsigned lds_off32(const void* p) {
  // flat shared address = {aperture_hi, lds_offset}: low 32 bits index LDS
  return (unsigned)(uintptr_t)p;
}

// Sense-reversing barrier across the NWG resident workgroups.
// bar[0] = arrive counter (returns to 0 after each round), bar[1] = generation.
__device__ __forceinline__ void grid_barrier(unsigned* bar) {
  __threadfence();
  __syncthreads();
  if (threadIdx.x == 0) {
    unsigned* cnt = bar;
    unsigned* gen = bar + 1;
    unsigned g = __atomic_load_n(gen, __ATOMIC_ACQUIRE);
    if (atomicAdd(cnt, 1u) == NWG - 1u) {
      __atomic_store_n(cnt, 0u, __ATOMIC_RELAXED);
      __atomic_store_n(gen, g + 1u, __ATOMIC_RELEASE);
    } else {
      while (__atomic_load_n(gen, __ATOMIC_ACQUIRE) == g) __builtin_amdgcn_s_sleep(1);
    }
  }
  __syncthreads();
}

// ---------------------------------------------------------------------------
// Kernel 1: l1 = mask(elu(x @ W1^T + b1)) stored as bf16 [B*T, H].
// One 16x16 output tile per wave; K=80 padded to 96 with zeros.
// Branch-free: indices are clamped in-bounds, values zeroed by select.
// Also resets the grid-barrier words (d_ws is poisoned by the harness).
// ---------------------------------------------------------------------------
__global__ void k_in_proj(const float* __restrict__ x, const float* __restrict__ W1,
                          const float* __restrict__ b1, const int* __restrict__ seq,
                          unsigned short* __restrict__ l1_u, unsigned* __restrict__ bar) {
  if (blockIdx.x == 0 && threadIdx.x == 0) { bar[0] = 0u; bar[1] = 0u; }
  __bf16* l1 = (__bf16*)l1_u;

  const int wave = threadIdx.x >> 5;
  const int lane = threadIdx.x & 31;
  const int tile = blockIdx.x * 8 + wave;       // 2000*32 = 64000 tiles total
  const int tm = tile >> 5;                     // M tile (rows of B*T)
  const int tn = tile & 31;                     // N tile (cols of H)
  const int hh = lane >> 4;                     // half-wave
  const int li = lane & 15;
  const int rowA = tm * 16 + li;                // A: lane holds row M=li
  const int n    = tn * 16 + li;                // B/C: lane holds col N=li

  v8f acc = {};
#pragma unroll
  for (int kst = 0; kst < 3; ++kst) {           // K = 80 -> 3 steps of 32 (padded)
    v16bf av, bv;
#pragma unroll
    for (int e = 0; e < 16; ++e) {
      // A operand (contiguous k order used here, same as B): lanes 0-15 K=0..15, lanes 16-31 K=16..31
      int kA = kst * 32 + 16 * hh + e;
      // B operand (contiguous k order used here, same as A): lanes 0-15 K=0..15, lanes 16-31 K=16..31
      int kB = kst * 32 + 16 * hh + e;
      int kAc = (kA < F_) ? kA : (F_ - 1);      // clamped: load always in-bounds
      int kBc = (kB < F_) ? kB : (F_ - 1);
      float va = x [(size_t)rowA * F_ + kAc];
      float vb = W1[(size_t)n    * F_ + kBc];
      av[e] = (__bf16)((kA < F_) ? va : 0.f);   // zero the K padding by select
      bv[e] = (__bf16)((kB < F_) ? vb : 0.f);
    }
    acc = wmma_bf16(av, bv, acc);
  }

  const float bn = b1[n];
#pragma unroll
  for (int r = 0; r < 8; ++r) {
    int row = tm * 16 + r + 8 * hh;             // C layout: M = r (+8 for upper half)
    int b = row / T_;
    int t = row - b * T_;
    float v = acc[r] + bn;
    v = (v > 0.f) ? v : (__expf(v) - 1.f);      // ELU
    if (t >= seq[b]) v = 0.f;                   // length mask
    l1[(size_t)row * H_ + n] = (__bf16)v;
  }
}

// ---------------------------------------------------------------------------
// Kernel 2: one LSTM layer, persistent scan over T with fused input projection.
//   g = [x_t | h_t] @ [w_ih ; w_hh]^T + b_ih + b_hh        (bf16 WMMA, f32 acc)
// WG `wg` owns hidden units [wg*32, wg*32+32) x 4 gates = 128 output columns.
// Weight slice (1024 x 128 bf16 = 256KB) lives in LDS for the whole scan.
// A-operands are staged with GLOBAL_LOAD_ASYNC_TO_LDS_B128 (ASYNCcnt).
// h is exchanged across the 16 WGs via a double-buffered global buffer.
// LDS: 256KB weights | 32KB A-staging (reused as gate buffer) | 4KB c | seq.
// ---------------------------------------------------------------------------
#define WREG_BYTES 262144
#define AREG_BYTES 32768
#define SMEM_BYTES (WREG_BYTES + AREG_BYTES + 4096 + 128)

__global__ void k_lstm_scan(const unsigned short* __restrict__ lin_u,
                            const float* __restrict__ w_ih, const float* __restrict__ w_hh,
                            const float* __restrict__ b_ih, const float* __restrict__ b_hh,
                            const int* __restrict__ seq,
                            unsigned short* __restrict__ hbuf_u,
                            unsigned short* __restrict__ out_bf_u,
                            float* __restrict__ out_f,
                            unsigned* __restrict__ bar) {
  extern __shared__ char smem[];
  char*  wreg  = smem;                                   // B-operand tiles, pre-swizzled
  char*  areg  = smem + WREG_BYTES;                      // A-operand tiles (one K-half)
  float* G     = (float*)areg;                           // gate buffer overlays areg
  float* c_lds = (float*)(smem + WREG_BYTES + AREG_BYTES);
  int*   seq_s = (int*)  (smem + WREG_BYTES + AREG_BYTES + 4096);

  const __bf16* lin   = (const __bf16*)lin_u;
  __bf16* hbuf        = (__bf16*)hbuf_u;                 // [2][B][H]
  __bf16* out_bf      = (__bf16*)out_bf_u;

  const int tid  = threadIdx.x;
  const int wg   = blockIdx.x;
  const int wave = tid >> 5;                             // = output N-tile (0..7)
  const int lane = tid & 31;
  const unsigned areg_base = lds_off32(areg);

  // --- init: cell state, seq cache, our slice of h(0) (parity 0) ---
  for (int j = tid; j < B_ * 32; j += TPB) c_lds[j] = 0.f;
  if (tid < B_) seq_s[tid] = seq[tid];
  for (int j = tid; j < 1024; j += TPB) {
    int m = j >> 5, jj = j & 31;
    hbuf[(size_t)m * H_ + wg * 32 + jj] = (__bf16)0.f;
  }

  // --- stage weight slice into LDS in WMMA B-operand order (fp32 -> bf16) ---
  // tile (nt, ks): lane holds col n = nt*16 + lane%16; element e -> K = ks*32 + 8*(lane/16) + (e<8 ? e : 8+e)
  for (int j = tid; j < 16384; j += TPB) {               // 16384 chunks of 8 elems
    int eg = j & 1, ln = (j >> 1) & 31, ks = (j >> 6) & 31, nt = (j >> 11) & 7;
    int hh = ln >> 4;
    int nl = nt * 16 + (ln & 15);                        // 0..127 within slice
    int gate = nl >> 5, jj = nl & 31;
    int ng = gate * H_ + wg * 32 + jj;                   // row in w_ih/w_hh [2048][512]
    int k  = ks * 32 + 8 * hh + eg * 16;                 // 8 consecutive K
    const float* src = (k < H_) ? (w_ih + (size_t)ng * H_ + k)
                                : (w_hh + (size_t)ng * H_ + (k - H_));
    __attribute__((aligned(16))) __bf16 tmp[8];
#pragma unroll
    for (int e = 0; e < 8; ++e) tmp[e] = (__bf16)src[e];
    *(float4*)(wreg + ((size_t)(nt * 32 + ks) * 32 + ln) * 32 + eg * 16) = *(float4*)tmp;
  }

  // per-wave bias for this lane's output column (same for all C rows)
  float bias_v;
  {
    int nl = wave * 16 + (lane & 15);
    int gate = nl >> 5, jj = nl & 31;
    int ng = gate * H_ + wg * 32 + jj;
    bias_v = b_ih[ng] + b_hh[ng];
  }
  __syncthreads();
  grid_barrier(bar);                                     // h(0) zeros visible everywhere

  // --- scan ---
  for (int t = 0; t < T_; ++t) {
    const int p = t & 1;                                 // read h from parity p
    v8f acc0, acc1;                                      // M-tiles (batch 0-15, 16-31)
#pragma unroll
    for (int r = 0; r < 8; ++r) { acc0[r] = bias_v; acc1[r] = bias_v; }

    // ---- K half 0: x_t (K = 0..511), async global->LDS, swizzled to A layout ----
    // j decode: m = j>>6 (row), kc = j&63 (16B chunk in row) -> coalesced reads.
    for (int j = tid; j < 2048; j += TPB) {
      int m  = j >> 6, kc = j & 63;
      int hh = kc & 1, eg = (kc >> 1) & 1, ks = kc >> 2;
      int mt = m >> 4, li = m & 15;
      const __bf16* gsrc = lin + ((size_t)m * T_ + t) * H_ + kc * 8;
      unsigned dst = areg_base + ((mt * 16 + ks) * 32 + hh * 16 + li) * 32 + eg * 16;
      async_b128_to_lds(dst, gsrc);
    }
    wait_asynccnt0();
    __syncthreads();
#pragma unroll 4
    for (int ks = 0; ks < 16; ++ks) {
      v16bf bv = *(const v16bf*)(wreg + ((size_t)(wave * 32 + ks) * 32 + lane) * 32);
      v16bf a0 = *(const v16bf*)(areg + (((0 * 16 + ks) * 32 + lane) * 32));
      v16bf a1 = *(const v16bf*)(areg + (((1 * 16 + ks) * 32 + lane) * 32));
      acc0 = wmma_bf16(a0, bv, acc0);
      acc1 = wmma_bf16(a1, bv, acc1);
    }
    __syncthreads();

    // ---- K half 1: h_t (K = 512..1023), async global->LDS ----
    for (int j = tid; j < 2048; j += TPB) {
      int m  = j >> 6, kc = j & 63;
      int hh = kc & 1, eg = (kc >> 1) & 1, ks = kc >> 2;
      int mt = m >> 4, li = m & 15;
      const __bf16* gsrc = hbuf + (size_t)p * B_ * H_ + (size_t)m * H_ + kc * 8;
      unsigned dst = areg_base + ((mt * 16 + ks) * 32 + hh * 16 + li) * 32 + eg * 16;
      async_b128_to_lds(dst, gsrc);
    }
    wait_asynccnt0();
    __syncthreads();
#pragma unroll 4
    for (int ks = 0; ks < 16; ++ks) {
      v16bf bv = *(const v16bf*)(wreg + ((size_t)(wave * 32 + 16 + ks) * 32 + lane) * 32);
      v16bf a0 = *(const v16bf*)(areg + (((0 * 16 + ks) * 32 + lane) * 32));
      v16bf a1 = *(const v16bf*)(areg + (((1 * 16 + ks) * 32 + lane) * 32));
      acc0 = wmma_bf16(a0, bv, acc0);
      acc1 = wmma_bf16(a1, bv, acc1);
    }
    __syncthreads();

    // ---- scatter gates to LDS (overlay on areg; row = gate*32+jj, stride 33) ----
    {
      int nl = wave * 16 + (lane & 15);
      int mh = 8 * (lane >> 4);
#pragma unroll
      for (int r = 0; r < 8; ++r) {
        G[nl * 33 + (r + mh)]      = acc0[r];            // batch rows 0..15
        G[nl * 33 + (16 + r + mh)] = acc1[r];            // batch rows 16..31
      }
    }
    __syncthreads();

    // ---- elementwise gate math, state update, outputs ----
    for (int u = tid; u < 1024; u += TPB) {              // u = m*32 + jj
      int m = u >> 5, jj = u & 31;
      float xi = G[(jj)      * 33 + m];
      float xf = G[(32 + jj) * 33 + m];
      float xg = G[(64 + jj) * 33 + m];
      float xo = G[(96 + jj) * 33 + m];
      float cp = c_lds[u];
      float cn = sigf(xf) * cp + sigf(xi) * tanhf(xg);
      float hn = sigf(xo) * tanhf(cn);
      c_lds[u] = cn;
      // unmasked h feeds the recurrence (matches lax.scan over full T)
      hbuf[(size_t)(1 - p) * B_ * H_ + (size_t)m * H_ + wg * 32 + jj] = (__bf16)hn;
      // masked h is the layer output
      float ov = (t < seq_s[m]) ? hn : 0.f;
      size_t oi = ((size_t)m * T_ + t) * H_ + wg * 32 + jj;
      if (out_f) out_f[oi] = ov; else out_bf[oi] = (__bf16)ov;
    }
    grid_barrier(bar);                                   // publish h(t+1) to all WGs
  }
}

// ---------------------------------------------------------------------------
extern "C" void kernel_launch(void* const* d_in, const int* in_sizes, int n_in,
                              void* d_out, int out_size, void* d_ws, size_t ws_size,
                              hipStream_t stream) {
  (void)in_sizes; (void)n_in; (void)out_size; (void)ws_size;
  const float* x     = (const float*)d_in[0];
  const int*   seq   = (const int*)  d_in[1];
  const float* W1    = (const float*)d_in[2];
  const float* b1    = (const float*)d_in[3];
  const float* w_ih1 = (const float*)d_in[4];
  const float* w_hh1 = (const float*)d_in[5];
  const float* b_ih1 = (const float*)d_in[6];
  const float* b_hh1 = (const float*)d_in[7];
  const float* w_ih2 = (const float*)d_in[8];
  const float* w_hh2 = (const float*)d_in[9];
  const float* b_ih2 = (const float*)d_in[10];
  const float* b_hh2 = (const float*)d_in[11];

  char* ws = (char*)d_ws;
  const size_t L1_BYTES   = (size_t)B_ * T_ * H_ * 2;    // 32,768,000 B (bf16)
  const size_t HBUF_BYTES = (size_t)2 * B_ * H_ * 2;     // 65,536 B
  unsigned short* l1   = (unsigned short*)(ws);
  unsigned short* lmid = (unsigned short*)(ws + L1_BYTES);
  unsigned short* hbuf = (unsigned short*)(ws + 2 * L1_BYTES);
  unsigned*       bar  = (unsigned*)      (ws + 2 * L1_BYTES + HBUF_BYTES);

  hipFuncSetAttribute(reinterpret_cast<const void*>(&k_lstm_scan),
                      hipFuncAttributeMaxDynamicSharedMemorySize, SMEM_BYTES);

  // l1 = mask(elu(x @ W1^T + b1))  -> bf16; also resets barrier state.
  k_in_proj<<<8000, TPB, 0, stream>>>(x, W1, b1, seq, l1, bar);
  // LSTM layer 1: l1 -> lmid (masked, bf16)
  k_lstm_scan<<<NWG, TPB, SMEM_BYTES, stream>>>(l1, w_ih1, w_hh1, b_ih1, b_hh1, seq,
                                                hbuf, lmid, nullptr, bar);
  // LSTM layer 2: lmid -> d_out (masked, fp32)
  k_lstm_scan<<<NWG, TPB, SMEM_BYTES, stream>>>(lmid, w_ih2, w_hh2, b_ih2, b_hh2, seq,
                                                hbuf, nullptr, (float*)d_out, bar);
}
